// BiGNN_Submanifold_52956946760190
// MI455X (gfx1250) — matrix operand order repaired and checked by reading, not compiled, not run on hardware
//
#include <hip/hip_runtime.h>

// ---------- types ----------
typedef __attribute__((ext_vector_type(16))) __bf16 v16bf;
typedef __attribute__((ext_vector_type(8)))  __bf16 v8bf;
typedef __attribute__((ext_vector_type(8)))  float   v8f;

#define EPSV 1e-3f

// ======================================================================
// feature cast: f32 [N,4] -> bf16 table [(N+1) x 32], cols>=4 zero, row N zero
// ======================================================================
__global__ void k_cast_feat0(const float* __restrict__ f, __bf16* __restrict__ tab, int n) {
    int i = blockIdx.x * blockDim.x + threadIdx.x;
    int total = (n + 1) * 32;
    if (i >= total) return;
    int row = i >> 5, c = i & 31;
    float v = (row < n && c < 4) ? f[row * 4 + c] : 0.f;
    tab[i] = (__bf16)v;
}

// ======================================================================
// weight fragment prep: w f32 [taps, cin, cout] -> B-fragments bf16
// frag[((t*ctiles+ct)*32 + lane)*16 + j], element K = (j<8 ? hi*8+j : 16+hi*8+j-8), col = ct*16+(l&15)
// zero-pads K >= cin (so reduction dims are padded to 32)
// ======================================================================
__global__ void k_prep_frag(const float* __restrict__ w, __bf16* __restrict__ frag,
                            int taps, int cin, int cout) {
    int ctiles = cout >> 4;
    int total = taps * ctiles * 32;
    int i = blockIdx.x * blockDim.x + threadIdx.x;
    if (i >= total) return;
    int l  = i & 31;
    int ct = (i >> 5) % ctiles;
    int t  = (i >> 5) / ctiles;
    int nn = ct * 16 + (l & 15);
    int hi = l >> 4;
    __bf16* dst = frag + (size_t)i * 16;
    for (int j = 0; j < 16; ++j) {
        int k = (j < 8) ? hi * 8 + j : 16 + hi * 8 + (j - 8);
        float v = (k < cin) ? w[((size_t)t * cin + k) * cout + nn] : 0.f;
        dst[j] = (__bf16)v;
    }
}

// ======================================================================
// zero a small float buffer (BN stats accumulator)
// ======================================================================
__global__ void k_zero(float* __restrict__ p, int n) {
    int i = blockIdx.x * blockDim.x + threadIdx.x;
    if (i < n) p[i] = 0.f;
}

// ======================================================================
// WMMA submanifold conv: 27-tap gather GEMM.
// One wave = 16 points x (CT*16) output channels. Feature table bf16 stride 32,
// row N is the zero sentinel. Weight B-fragments staged in LDS.
// ======================================================================
template <int CT>
__global__ __launch_bounds__(128)
void k_conv_wmma(const __bf16* __restrict__ feat, const int* __restrict__ nbr,
                 const __bf16* __restrict__ frag, float* __restrict__ y, int n) {
    __shared__ __align__(16) __bf16 sfrag[27 * CT * 512];
    const int tid = threadIdx.x;
    {   // cooperative stage of all B fragments into LDS (27*CT*1KB)
        const uint4* s = (const uint4*)frag;
        uint4* d = (uint4*)sfrag;
        const int chunks = CT * 1728;           // 27*CT*512*2 / 16
        for (int i = tid; i < chunks; i += 128) d[i] = s[i];
    }
    __syncthreads();

    const int ntiles = (n + 15) >> 4;
    const int wave = blockIdx.x * 4 + (tid >> 5);
    if (wave >= ntiles) return;
    const int l = tid & 31, m = l & 15, hi = l >> 4;
    const int n0 = wave << 4;
    int pr = n0 + m; if (pr >= n) pr = n - 1;

    const v8f zz = {};
    v8f c[CT];
#pragma unroll
    for (int ct = 0; ct < CT; ++ct) c[ct] = zz;

    const int* nrow = nbr + (size_t)pr * 27;
    for (int t = 0; t < 27; ++t) {
        int row = nrow[t];                       // sentinel n -> zero row
        const __bf16* fp = feat + ((size_t)row << 5) + hi * 8;
        union { v16bf v; v8bf h[2]; } a;
        a.h[0] = *(const v8bf*)fp;               // K = hi*8 .. hi*8+7
        a.h[1] = *(const v8bf*)(fp + 16);        // K = 16+hi*8 .. 16+hi*8+7
#pragma unroll
        for (int ct = 0; ct < CT; ++ct) {
            const v16bf b = *(const v16bf*)(sfrag + ((t * CT + ct) * 32 + l) * 16);
            c[ct] = __builtin_amdgcn_wmma_f32_16x16x32_bf16(
                false, a.v, false, b, (short)0, c[ct], false, false);
        }
    }
    const int cout = CT * 16;
#pragma unroll
    for (int ct = 0; ct < CT; ++ct)
#pragma unroll
        for (int r = 0; r < 8; ++r) {
            int row = n0 + hi * 8 + r;
            if (row < n) y[(size_t)row * cout + ct * 16 + m] = c[ct][r];
        }
}

// ======================================================================
// per-channel sum / sumsq for BN: atomics via LDS then global
// ======================================================================
__global__ __launch_bounds__(256)
void k_stats(const float* __restrict__ y, float* __restrict__ stats, int n, int C) {
    __shared__ float sh[128];
    int t = threadIdx.x;
    if (t < 2 * C) sh[t] = 0.f;
    __syncthreads();
    int rpb = 256 / C;
    int c = t % C, ro = t / C;
    float s = 0.f, q = 0.f;
    for (int r = blockIdx.x * rpb + ro; r < n; r += gridDim.x * rpb) {
        float v = y[(size_t)r * C + c];
        s += v; q += v * v;
    }
    atomicAdd(&sh[c], s);
    atomicAdd(&sh[C + c], q);
    __syncthreads();
    if (t < 2 * C) atomicAdd(&stats[t], sh[t]);
}

// ======================================================================
// BN + ReLU + cast to bf16 table [(N+1) x 32] (pad cols & sentinel row zero);
// optionally mirror first 32 cols into lrx[:,0:32] (for r3)
// ======================================================================
__global__ void k_bn_cast(const float* __restrict__ y, const float* __restrict__ stats,
                          const float* __restrict__ gamma, const float* __restrict__ beta,
                          __bf16* __restrict__ tab, __bf16* __restrict__ lrx, int n, int C) {
    int i = blockIdx.x * blockDim.x + threadIdx.x;
    int total = (n + 1) * 32;
    if (i >= total) return;
    int row = i >> 5, c = i & 31;
    float v = 0.f;
    if (row < n && c < C) {
        float inv = 1.f / (float)n;
        float mean = stats[c] * inv;
        float var  = stats[C + c] * inv - mean * mean;
        float x = (y[(size_t)row * C + c] - mean) * rsqrtf(var + EPSV) * gamma[c] + beta[c];
        v = fmaxf(x, 0.f);
    }
    tab[i] = (__bf16)v;
    if (lrx != nullptr && row < n) lrx[(size_t)row * 224 + c] = (__bf16)v;
}

// ======================================================================
// grouper: one wave per (point, g, s). A = 16 gathered rows x 32ch,
// B = wg[g,s] (2 col tiles), h = relu(A@B + bias), max over the 16 rows
// (8 accum VGPRs per lane + shfl_xor(16)), write bf16 into lrx columns.
// ======================================================================
__global__ __launch_bounds__(128)
void k_grouper(const __bf16* __restrict__ r1, const __bf16* __restrict__ r2,
               const __bf16* __restrict__ r3, const int* __restrict__ grp,
               const __bf16* __restrict__ frag, const float* __restrict__ bg,
               __bf16* __restrict__ lrx, int n) {
    __shared__ __align__(16) __bf16 sfrag[6 * 2 * 512];
    const int tid = threadIdx.x;
    {   const uint4* s = (const uint4*)frag;
        uint4* d = (uint4*)sfrag;
        for (int i = tid; i < 768; i += 128) d[i] = s[i];
    }
    __syncthreads();

    long wave = (long)blockIdx.x * 4 + (tid >> 5);
    if (wave >= (long)n * 6) return;
    int gs = (int)(wave / n);           // 0..5  (g = gs/3, s = gs%3)
    int pt = (int)(wave % n);
    int sc = gs % 3;
    const __bf16* rt = (sc == 0) ? r1 : (sc == 1) ? r2 : r3;

    const int l = tid & 31, m = l & 15, hi = l >> 4;
    int row = grp[((size_t)gs * n + pt) * 16 + m];
    const __bf16* fp = rt + ((size_t)row << 5) + hi * 8;
    union { v16bf v; v8bf h[2]; } a;
    a.h[0] = *(const v8bf*)fp;
    a.h[1] = *(const v8bf*)(fp + 16);

    const v8f zz = {};
#pragma unroll
    for (int ct = 0; ct < 2; ++ct) {
        const v16bf b = *(const v16bf*)(sfrag + ((gs * 2 + ct) * 32 + l) * 16);
        v8f c = zz;
        c = __builtin_amdgcn_wmma_f32_16x16x32_bf16(
            false, a.v, false, b, (short)0, c, false, false);
        float bias = bg[gs * 32 + ct * 16 + m];
        float mx = 0.f;                              // relu floor
#pragma unroll
        for (int r = 0; r < 8; ++r) mx = fmaxf(mx, c[r] + bias);
        mx = fmaxf(mx, __shfl_xor(mx, 16, 32));      // combine M and M+8 halves
        if (hi == 0)
            lrx[(size_t)pt * 224 + 32 + gs * 32 + ct * 16 + m] = (__bf16)mx;
    }
}

// ======================================================================
// final GEMM: lrx [N,224] bf16 @ w_out frag -> y [N,64] f32.
// One wave = 16 rows x 64 cols: 7 K-steps x 4 col tiles = 28 WMMAs.
// ======================================================================
__global__ __launch_bounds__(128)
void k_final_gemm(const __bf16* __restrict__ lrx, const __bf16* __restrict__ frag,
                  float* __restrict__ y, int n) {
    __shared__ __align__(16) __bf16 sfrag[7 * 4 * 512];
    const int tid = threadIdx.x;
    {   const uint4* s = (const uint4*)frag;
        uint4* d = (uint4*)sfrag;
        for (int i = tid; i < 1792; i += 128) d[i] = s[i];
    }
    __syncthreads();

    const int ntiles = (n + 15) >> 4;
    const int wave = blockIdx.x * 4 + (tid >> 5);
    if (wave >= ntiles) return;
    const int l = tid & 31, m = l & 15, hi = l >> 4;
    const int n0 = wave << 4;
    int pr = n0 + m; if (pr >= n) pr = n - 1;

    const v8f zz = {};
    v8f c[4];
#pragma unroll
    for (int i = 0; i < 4; ++i) c[i] = zz;

    const __bf16* arow = lrx + (size_t)pr * 224;
    for (int kb = 0; kb < 7; ++kb) {
        const __bf16* fp = arow + kb * 32 + hi * 8;
        union { v16bf v; v8bf h[2]; } a;
        a.h[0] = *(const v8bf*)fp;
        a.h[1] = *(const v8bf*)(fp + 16);
#pragma unroll
        for (int ct = 0; ct < 4; ++ct) {
            const v16bf b = *(const v16bf*)(sfrag + ((kb * 4 + ct) * 32 + l) * 16);
            c[ct] = __builtin_amdgcn_wmma_f32_16x16x32_bf16(
                false, a.v, false, b, (short)0, c[ct], false, false);
        }
    }
#pragma unroll
    for (int ct = 0; ct < 4; ++ct)
#pragma unroll
        for (int r = 0; r < 8; ++r) {
            int row = n0 + hi * 8 + r;
            if (row < n) y[(size_t)row * 64 + ct * 16 + m] = c[ct][r];
        }
}

// ======================================================================
// final BN + ReLU -> f32 output [N,64]
// ======================================================================
__global__ void k_bn_out(const float* __restrict__ y, const float* __restrict__ stats,
                         const float* __restrict__ gamma, const float* __restrict__ beta,
                         float* __restrict__ out, int n) {
    int i = blockIdx.x * blockDim.x + threadIdx.x;
    if (i >= n * 64) return;
    int c = i & 63;
    float inv = 1.f / (float)n;
    float mean = stats[c] * inv;
    float var  = stats[64 + c] * inv - mean * mean;
    float x = (y[i] - mean) * rsqrtf(var + EPSV) * gamma[c] + beta[c];
    out[i] = fmaxf(x, 0.f);
}

// ======================================================================
// host-side launcher
// ======================================================================
extern "C" void kernel_launch(void* const* d_in, const int* in_sizes, int n_in,
                              void* d_out, int out_size, void* d_ws, size_t ws_size,
                              hipStream_t stream) {
    const float* features = (const float*)d_in[0];
    const int*   nbr      = (const int*)  d_in[1];
    const int*   grp      = (const int*)  d_in[2];
    const float* w_in = (const float*)d_in[3];
    const float* g_in = (const float*)d_in[4];
    const float* b_in = (const float*)d_in[5];
    const float* w1   = (const float*)d_in[6];
    const float* g1   = (const float*)d_in[7];
    const float* b1   = (const float*)d_in[8];
    const float* w2   = (const float*)d_in[9];
    const float* g2   = (const float*)d_in[10];
    const float* b2   = (const float*)d_in[11];
    const float* w3   = (const float*)d_in[12];
    const float* g3   = (const float*)d_in[13];
    const float* b3   = (const float*)d_in[14];
    const float* wg   = (const float*)d_in[15];
    const float* bg   = (const float*)d_in[16];
    const float* wout = (const float*)d_in[17];
    const float* gout = (const float*)d_in[18];
    const float* bout = (const float*)d_in[19];

    const int n = in_sizes[0] / 4;            // N = 120000
    const int ntiles = (n + 15) / 16;

    // ---- workspace carve ----
    char* base = (char*)d_ws;
    size_t off = 0;
    auto carve = [&](size_t bytes) -> void* {
        void* p = base + off;
        off += (bytes + 255) & ~(size_t)255;
        return p;
    };
    const size_t tabB = (size_t)(n + 1) * 32 * sizeof(__bf16);
    __bf16* t_io = (__bf16*)carve(tabB);          // feat0, then x16 (sequential reuse)
    __bf16* t_r1 = (__bf16*)carve(tabB);
    __bf16* t_r2 = (__bf16*)carve(tabB);
    __bf16* t_r3 = (__bf16*)carve(tabB);
    float*  ybuf = (float*)carve((size_t)n * 64 * sizeof(float));
    __bf16* lrx  = (__bf16*)carve((size_t)n * 224 * sizeof(__bf16));
    float*  stats = (float*)carve(512);
    __bf16* f_win = (__bf16*)carve((size_t)27 * 1 * 512 * 2);
    __bf16* f_w1  = (__bf16*)carve((size_t)27 * 2 * 512 * 2);
    __bf16* f_w2  = (__bf16*)carve((size_t)27 * 2 * 512 * 2);
    __bf16* f_w3  = (__bf16*)carve((size_t)27 * 2 * 512 * 2);
    __bf16* f_wg  = (__bf16*)carve((size_t)6 * 2 * 512 * 2);
    __bf16* f_wo  = (__bf16*)carve((size_t)7 * 4 * 512 * 2);
    (void)ws_size; (void)n_in; (void)out_size;

    const int elemTab = (n + 1) * 32;
    const int gTab = (elemTab + 255) / 256;

    // feature table + weight fragment prep
    k_cast_feat0<<<gTab, 256, 0, stream>>>(features, t_io, n);
    k_prep_frag<<<(27 * 1 * 32 + 255) / 256, 256, 0, stream>>>(w_in, f_win, 27, 4, 16);
    k_prep_frag<<<(27 * 2 * 32 + 255) / 256, 256, 0, stream>>>(w1, f_w1, 27, 16, 32);
    k_prep_frag<<<(27 * 2 * 32 + 255) / 256, 256, 0, stream>>>(w2, f_w2, 27, 32, 32);
    k_prep_frag<<<(27 * 2 * 32 + 255) / 256, 256, 0, stream>>>(w3, f_w3, 27, 32, 32);
    k_prep_frag<<<(6 * 2 * 32 + 255) / 256, 256, 0, stream>>>(wg, f_wg, 6, 32, 32);
    k_prep_frag<<<(7 * 4 * 32 + 255) / 256, 256, 0, stream>>>(wout, f_wo, 7, 32, 64);

    const int gConv = (ntiles + 3) / 4;

    // layer in: 4(pad32) -> 16
    k_conv_wmma<1><<<gConv, 128, 0, stream>>>(t_io, nbr, f_win, ybuf, n);
    k_zero<<<1, 128, 0, stream>>>(stats, 128);
    k_stats<<<512, 256, 0, stream>>>(ybuf, stats, n, 16);
    k_bn_cast<<<gTab, 256, 0, stream>>>(ybuf, stats, g_in, b_in, t_io, nullptr, n, 16);

    // layer 1: 16(pad32) -> 32
    k_conv_wmma<2><<<gConv, 128, 0, stream>>>(t_io, nbr, f_w1, ybuf, n);
    k_zero<<<1, 128, 0, stream>>>(stats, 128);
    k_stats<<<512, 256, 0, stream>>>(ybuf, stats, n, 32);
    k_bn_cast<<<gTab, 256, 0, stream>>>(ybuf, stats, g1, b1, t_r1, nullptr, n, 32);

    // layer 2: 32 -> 32
    k_conv_wmma<2><<<gConv, 128, 0, stream>>>(t_r1, nbr, f_w2, ybuf, n);
    k_zero<<<1, 128, 0, stream>>>(stats, 128);
    k_stats<<<512, 256, 0, stream>>>(ybuf, stats, n, 32);
    k_bn_cast<<<gTab, 256, 0, stream>>>(ybuf, stats, g2, b2, t_r2, nullptr, n, 32);

    // layer 3: 32 -> 32 (also mirror r3 into lrx[:,0:32])
    k_conv_wmma<2><<<gConv, 128, 0, stream>>>(t_r2, nbr, f_w3, ybuf, n);
    k_zero<<<1, 128, 0, stream>>>(stats, 128);
    k_stats<<<512, 256, 0, stream>>>(ybuf, stats, n, 32);
    k_bn_cast<<<gTab, 256, 0, stream>>>(ybuf, stats, g3, b3, t_r3, lrx, n, 32);

    // groupers: N*6 waves, 4 waves per block
    {
        long waves = (long)n * 6;
        int blocks = (int)((waves + 3) / 4);
        k_grouper<<<blocks, 128, 0, stream>>>(t_r1, t_r2, t_r3, grp, f_wg, bg, lrx, n);
    }

    // final GEMM [N,224]x[224,64] + BN + ReLU
    k_final_gemm<<<gConv, 128, 0, stream>>>(lrx, f_wo, ybuf, n);
    k_zero<<<1, 128, 0, stream>>>(stats, 128);
    k_stats<<<512, 256, 0, stream>>>(ybuf, stats, n, 64);
    k_bn_out<<<(n * 64 + 255) / 256, 256, 0, stream>>>(ybuf, stats, gout, bout,
                                                       (float*)d_out, n);
}
